// GGNNRNN_66254165508932
// MI455X (gfx1250) — matrix operand order repaired and checked by reading, not compiled, not run on hardware
//
#include <hip/hip_runtime.h>
#include <hip/hip_bf16.h>
#include <math.h>

// Toggle: stage the GEMM B panel with CDNA5 async global->LDS copies
// (global_load_async_to_lds_b128 + s_wait_asynccnt) instead of a synchronous
// global->VGPR->ds_store copy.
#define ASYNC_B_STAGE 1

// ---------------------------------------------------------------------------
// Types for CDNA5 WMMA (wave32): v_wmma_f32_16x16x32_bf16
// ---------------------------------------------------------------------------
typedef __attribute__((ext_vector_type(16))) __bf16 bf16x16;
typedef __attribute__((ext_vector_type(8)))  float  f32x8;

__device__ __forceinline__ unsigned short f2bf(float f) {
  union { float f; unsigned u; } v; v.f = f;
  unsigned r = v.u + 0x7FFFu + ((v.u >> 16) & 1u);   // round-to-nearest-even
  return (unsigned short)(r >> 16);
}

__device__ __forceinline__ float sigmoidf_(float x) { return 1.f / (1.f + expf(-x)); }

// ---------------------------------------------------------------------------
// Pack a weight matrix into WMMA B-fragment order (bf16).
// Logical B is K x N (row-major).  If transpose!=0, src is (N,K) row-major and
// B[k][n] = src[n*K + k]  (i.e. implements  A @ src.T).
// Fragment layout (v_wmma_*_16x16x32_bf16 B operand, 32x16 tile):
//   lanes 0-15  : column n, K = kb+0..15  (2 per VGPR, K-consecutive)
//   lanes 16-31 : column n, K = kb+16..31
// Each lane's 16 bf16 values contiguous (32B).
// dst element index = (((kt*ntiles + nt)*32 + lane)*16 + t)
// ---------------------------------------------------------------------------
__global__ void pack_b_kernel(const float* __restrict__ src,
                              unsigned short* __restrict__ dst,
                              int K, int N, int transpose) {
  size_t idx = (size_t)blockIdx.x * blockDim.x + threadIdx.x;
  size_t total = (size_t)K * N;
  if (idx >= total) return;
  int t    = (int)(idx & 15);
  int lane = (int)((idx >> 4) & 31);
  size_t tile = idx >> 9;
  int ntiles = N >> 4;
  int nt = (int)(tile % ntiles);
  int kt = (int)(tile / ntiles);
  int k = kt * 32 + ((lane >> 4) << 4) + t;
  int n = nt * 16 + (lane & 15);
  float v = transpose ? src[(size_t)n * K + k] : src[(size_t)k * N + n];
  dst[idx] = f2bf(v);
}

// ---------------------------------------------------------------------------
// GEMM: C[M x N] (f32, row-major) = A[M x KDIM] x Bp (pre-packed bf16
// fragments).  N % 64 == 0.  A is bf16 row-major if ABF16 (two b128 loads per
// fragment, no conversion VALU), else f32 converted to bf16 on load.
// 8 waves/block; block tile 128(M) x 64(N); wave tile 16(M) x 64(N).
// B panel for the block (KDIM x 64 = 32KB bf16 @ KDIM=256) staged in LDS once
// via async global->LDS DMA; fragments then come from ds_load_b128.
// K-loop fully unrolled -> 4*KT wmma.
// ---------------------------------------------------------------------------
template <int KDIM, bool ABF16>
__global__ __launch_bounds__(256)
void gemm_bf16_wmma(const void* __restrict__ Av,
                    const unsigned short* __restrict__ Bp,
                    float* __restrict__ C,
                    int M, int N) {
  constexpr int KT = KDIM / 32;
  __shared__ uint4 sB4[KT * 256];       // 32KB @ KT=8

  const int tid  = threadIdx.x;
  const int lane = tid & 31;
  const int wave = tid >> 5;
  const int half = lane >> 4;           // 0: lanes 0-15, 1: lanes 16-31
  const int l16  = lane & 15;

  const int mBase  = blockIdx.y * 128 + wave * 16;
  const int nBase  = blockIdx.x * 64;
  const int ntiles = N >> 4;
  const int nt0    = nBase >> 4;

  // ---- stage this block's B panel into LDS.
#if ASYNC_B_STAGE
  {
    // Per kt: 4 tiles * 1KB contiguous in Bp; lds dst stride 4KB per kt.
    // global_load_async_to_lds_b128: VDST = per-lane LDS byte address
    // (flat shared-pointer low 32 bits == wave-relative LDS address),
    // VADDR = 64-bit global address, tracked by ASYNCcnt.
    const char* gsrc = (const char*)Bp + (size_t)nt0 * 1024 + (size_t)tid * 16;
    const size_t gstride = (size_t)ntiles * 1024;
#pragma unroll
    for (int kt = 0; kt < KT; ++kt) {
      unsigned ldsAddr = (unsigned)(size_t)(sB4 + kt * 256 + tid);
      const char* g = gsrc + (size_t)kt * gstride;
      asm volatile("global_load_async_to_lds_b128 %0, %1, off"
                   :: "v"(ldsAddr), "v"(g) : "memory");
    }
    // Each wave drains its own async copies before the workgroup barrier.
    asm volatile("s_wait_asynccnt 0x0" ::: "memory");
  }
#else
  {
    const uint4* bsrc = (const uint4*)Bp;
#pragma unroll
    for (int kt = 0; kt < KT; ++kt)
      sB4[kt * 256 + tid] = bsrc[((size_t)kt * ntiles + nt0) * 64 + tid];
  }
#endif
  __syncthreads();
  const unsigned short* sBu = (const unsigned short*)sB4;

  // A-fragment row for this lane (clamped so OOB tiles still load safely;
  // stores are guarded below).
  int row = mBase + l16;
  if (row >= M) row = M - 1;
  const float*          arowf = (const float*)Av          + (ABF16 ? 0 : (size_t)row * KDIM);
  const unsigned short* arowh = (const unsigned short*)Av + (ABF16 ? (size_t)row * KDIM : 0);

  f32x8 acc0 = {}, acc1 = {}, acc2 = {}, acc3 = {};

#pragma unroll
  for (int kt = 0; kt < KT; ++kt) {
    const int kb = kt * 32;
    // ---- A fragment: 16x32, contiguous-k order used by this code (not the ISA table order).
    // lane half h holds K = kb+16h+{0..15} (contiguous, matching packed B).
    union { bf16x16 v; unsigned short u[16]; uint4 q[2]; } au;
    if (ABF16) {
      const unsigned short* p0 = arowh + kb + half * 16;
      au.q[0] = *(const uint4*)(p0);
      au.q[1] = *(const uint4*)(p0 + 8);
    } else {
      const float* p0 = arowf + kb + half * 16;
      float tmp[16];
      *(float4*)(tmp + 0)  = *(const float4*)(p0 + 0);
      *(float4*)(tmp + 4)  = *(const float4*)(p0 + 4);
      *(float4*)(tmp + 8)  = *(const float4*)(p0 + 8);
      *(float4*)(tmp + 12) = *(const float4*)(p0 + 12);
#pragma unroll
      for (int t = 0; t < 16; ++t) au.u[t] = f2bf(tmp[t]);
    }
    const unsigned short* bt = sBu + kt * 2048 + lane * 16;
    bf16x16 b0 = *(const bf16x16*)(bt + 0 * 512);
    bf16x16 b1 = *(const bf16x16*)(bt + 1 * 512);
    bf16x16 b2 = *(const bf16x16*)(bt + 2 * 512);
    bf16x16 b3 = *(const bf16x16*)(bt + 3 * 512);

    acc0 = __builtin_amdgcn_wmma_f32_16x16x32_bf16(false, au.v, false, b0, (short)0, acc0, false, false);
    acc1 = __builtin_amdgcn_wmma_f32_16x16x32_bf16(false, au.v, false, b1, (short)0, acc1, false, false);
    acc2 = __builtin_amdgcn_wmma_f32_16x16x32_bf16(false, au.v, false, b2, (short)0, acc2, false, false);
    acc3 = __builtin_amdgcn_wmma_f32_16x16x32_bf16(false, au.v, false, b3, (short)0, acc3, false, false);
  }

  // ---- C store. Lane layout: VGPR v -> row crow0+v, col nBase + j*16 + l16.
  // Fast path (wave-uniform): whole 16-row tile in bounds -> unguarded stores
  // from one base pointer with immediate column offsets.
  const int crow0 = mBase + half * 8;
  if (mBase + 16 <= M) {
    float* cptr = C + (size_t)crow0 * N + (nBase + l16);
#pragma unroll
    for (int v = 0; v < 8; ++v) {
      cptr[0]  = acc0[v];
      cptr[16] = acc1[v];
      cptr[32] = acc2[v];
      cptr[48] = acc3[v];
      cptr += N;
    }
  } else {
    float* cptr = C + (size_t)crow0 * N + (nBase + l16);
#pragma unroll
    for (int v = 0; v < 8; ++v) {
      if (crow0 + v < M) {
        cptr[0]  = acc0[v];
        cptr[16] = acc1[v];
        cptr[32] = acc2[v];
        cptr[48] = acc3[v];
      }
      cptr += N;
    }
  }
}

// ---------------------------------------------------------------------------
// Embedding gather: h[n][d] = embed_w[x[n]][d]  (f32 + bf16 copy)
// ---------------------------------------------------------------------------
__global__ void embed_kernel(const int* __restrict__ x,
                             const float* __restrict__ embed_w,
                             float* __restrict__ h,
                             unsigned short* __restrict__ hb, int NN) {
  size_t idx = (size_t)blockIdx.x * blockDim.x + threadIdx.x;
  if (idx >= (size_t)NN * 256) return;
  int n = (int)(idx >> 8);
  int d = (int)(idx & 255);
  float v = embed_w[(size_t)x[n] * 256 + d];
  h[idx]  = v;
  hb[idx] = f2bf(v);
}

// ---------------------------------------------------------------------------
// Edge scatter-add: agg[tgt[e]] += m[src[e]]   (float4 gather + 4 f32 atomics)
// One thread per (edge, 4-dim group); 64 groups per edge.
// ---------------------------------------------------------------------------
__global__ void scatter_add_kernel(const float* __restrict__ m,
                                   const int* __restrict__ src,
                                   const int* __restrict__ tgt,
                                   float* __restrict__ agg, int NE) {
  size_t idx = (size_t)blockIdx.x * blockDim.x + threadIdx.x;
  int e = (int)(idx >> 6);
  if (e >= NE) return;
  int q = (int)(idx & 63) * 4;
  int s = src[e], t = tgt[e];
  float4 v = *(const float4*)(m + (size_t)s * 256 + q);
  float* dst = agg + (size_t)t * 256 + q;
  atomicAdd(dst + 0, v.x);
  atomicAdd(dst + 1, v.y);
  atomicAdd(dst + 2, v.z);
  atomicAdd(dst + 3, v.w);
}

// ---------------------------------------------------------------------------
// GRU update for a node chunk. gi/gh are (chunkNodes x 768) local buffers.
// Writes both the f32 state and its bf16 mirror.
// ---------------------------------------------------------------------------
__global__ void gru_update_kernel(const float* __restrict__ gi,
                                  const float* __restrict__ gh,
                                  const float* __restrict__ b_ih,
                                  const float* __restrict__ b_hh,
                                  float* __restrict__ h,
                                  unsigned short* __restrict__ hb,
                                  int nodeBase, int nNodes) {
  size_t idx = (size_t)blockIdx.x * blockDim.x + threadIdx.x;
  if (idx >= (size_t)nNodes * 256) return;
  int local = (int)(idx >> 8);
  int d = (int)(idx & 255);
  size_t gb = (size_t)local * 768;
  float i_r = gi[gb + d]       + b_ih[d];
  float i_z = gi[gb + 256 + d] + b_ih[256 + d];
  float i_n = gi[gb + 512 + d] + b_ih[512 + d];
  float h_r = gh[gb + d]       + b_hh[d];
  float h_z = gh[gb + 256 + d] + b_hh[256 + d];
  float h_n = gh[gb + 512 + d] + b_hh[512 + d];
  size_t hi = (size_t)(nodeBase + local) * 256 + d;
  float hold = h[hi];
  float r = sigmoidf_(i_r + h_r);
  float z = sigmoidf_(i_z + h_z);
  float nn = tanhf(i_n + r * h_n);
  float hnew = (1.f - z) * nn + z * hold;
  h[hi]  = hnew;
  hb[hi] = f2bf(hnew);
}

// ---------------------------------------------------------------------------
// Gate pass 1: per node, wn = exp(sigmoid(h.gate_w + b)); denom += wn.
// (softmax folded: hg = sum(wn*h)/sum(wn); sig in (0,1) so exp is safe.)
// ---------------------------------------------------------------------------
__global__ void gate_node_kernel(const float* __restrict__ h,
                                 const float* __restrict__ gate_w,
                                 const float* __restrict__ gate_b,
                                 float* __restrict__ wn,
                                 float* __restrict__ denom, int NN) {
  int wave = threadIdx.x >> 5, lane = threadIdx.x & 31;
  int node = blockIdx.x * 8 + wave;
  if (node >= NN) return;
  const float* hr = h + (size_t)node * 256;
  float s = 0.f;
#pragma unroll
  for (int t = 0; t < 8; ++t) { int d = lane + 32 * t; s += hr[d] * gate_w[d]; }
#pragma unroll
  for (int o = 16; o > 0; o >>= 1) s += __shfl_down(s, o);
  if (lane == 0) {
    float w = expf(sigmoidf_(s + gate_b[0]));
    wn[node] = w;
    atomicAdd(denom, w);
  }
}

// Gate pass 2: hg[d] += sum over node range of wn[n]*h[n][d]
__global__ void gate_accum_kernel(const float* __restrict__ h,
                                  const float* __restrict__ wn,
                                  float* __restrict__ hg, int NN) {
  int d = threadIdx.x;
  int per = (NN + gridDim.x - 1) / (int)gridDim.x;
  int start = blockIdx.x * per;
  int end = start + per; if (end > NN) end = NN;
  float acc = 0.f;
  for (int n = start; n < end; ++n) acc += wn[n] * h[(size_t)n * 256 + d];
  atomicAdd(&hg[d], acc);
}

// LSTM pre-activations: g[row] = (hg/denom) . W_ih[row] + b_ih[row] + b_hh[row]
__global__ void lstm_matvec_kernel(const float* __restrict__ W,
                                   const float* __restrict__ hg,
                                   const float* __restrict__ denom,
                                   const float* __restrict__ b_ih,
                                   const float* __restrict__ b_hh,
                                   float* __restrict__ g) {
  int wave = threadIdx.x >> 5, lane = threadIdx.x & 31;
  int row = blockIdx.x * 8 + wave;
  if (row >= 2048) return;
  const float* wr = W + (size_t)row * 256;
  float s = 0.f;
#pragma unroll
  for (int t = 0; t < 8; ++t) { int d = lane + 32 * t; s += wr[d] * hg[d]; }
#pragma unroll
  for (int o = 16; o > 0; o >>= 1) s += __shfl_down(s, o);
  if (lane == 0) g[row] = s * (1.f / denom[0]) + b_ih[row] + b_hh[row];
}

// LSTM gates -> h1 (c0 = 0 so forget gate is irrelevant)
__global__ void lstm_act_kernel(const float* __restrict__ g, float* __restrict__ h1) {
  int j = threadIdx.x;            // 512 threads, 1 block
  float ig = g[j];
  float gg = g[1024 + j];
  float og = g[1536 + j];
  float c1 = sigmoidf_(ig) * tanhf(gg);
  h1[j] = sigmoidf_(og) * tanhf(c1);
}

// logits[v] = dense_w[v] . h1 + dense_b[v]
__global__ void dense_kernel(const float* __restrict__ W,
                             const float* __restrict__ b,
                             const float* __restrict__ h1,
                             float* __restrict__ out, int NV) {
  __shared__ float sh[512];
  sh[threadIdx.x] = h1[threadIdx.x];
  sh[threadIdx.x + 256] = h1[threadIdx.x + 256];
  __syncthreads();
  int wave = threadIdx.x >> 5, lane = threadIdx.x & 31;
  int row = blockIdx.x * 8 + wave;
  if (row >= NV) return;
  const float* wr = W + (size_t)row * 512;
  float s = 0.f;
#pragma unroll
  for (int t = 0; t < 16; ++t) { int d = lane + 32 * t; s += wr[d] * sh[d]; }
#pragma unroll
  for (int o = 16; o > 0; o >>= 1) s += __shfl_down(s, o);
  if (lane == 0) out[row] = s + b[row];
}

// ---------------------------------------------------------------------------
extern "C" void kernel_launch(void* const* d_in, const int* in_sizes, int n_in,
                              void* d_out, int out_size, void* d_ws, size_t ws_size,
                              hipStream_t stream) {
  (void)n_in; (void)out_size; (void)ws_size;

  const int*   x          = (const int*)d_in[0];
  const int*   edge_index = (const int*)d_in[1];
  const float* embed_w    = (const float*)d_in[3];
  const float* ggnn_w     = (const float*)d_in[5];
  const float* gru_w_ih   = (const float*)d_in[6];
  const float* gru_w_hh   = (const float*)d_in[7];
  const float* gru_b_ih   = (const float*)d_in[8];
  const float* gru_b_hh   = (const float*)d_in[9];
  const float* gate_w     = (const float*)d_in[10];
  const float* gate_b     = (const float*)d_in[11];
  const float* lstm_w_ih  = (const float*)d_in[12];
  const float* lstm_b_ih  = (const float*)d_in[14];
  const float* lstm_b_hh  = (const float*)d_in[15];
  const float* dense_w    = (const float*)d_in[16];
  const float* dense_b    = (const float*)d_in[17];
  float* out = (float*)d_out;

  const int NN = in_sizes[0];          // 200000 nodes
  const int NE = in_sizes[1] / 2;      // 1600000 edges
  const int D = 256, G = 768;
  const int NV = in_sizes[17];         // 50257
  const int CHUNK = 25000;

  // -------- workspace carve-out (~875 MB) --------
  char* ws = (char*)d_ws;
  size_t off = 0;
  auto carve = [&](size_t bytes) -> void* {
    void* p = ws + off;
    off = (off + bytes + 255) & ~(size_t)255;
    return p;
  };
  float*          h     = (float*)carve((size_t)NN * D * 4);
  unsigned short* hb    = (unsigned short*)carve((size_t)NN * D * 2);
  float*          m     = (float*)carve((size_t)NN * D * 4);
  float*          agg   = (float*)carve((size_t)NN * D * 4);
  float*          gi    = (float*)carve((size_t)CHUNK * G * 4);
  float*          gh    = (float*)carve((size_t)CHUNK * G * 4);
  unsigned short* BpG   = (unsigned short*)carve((size_t)4 * D * D * 2);
  unsigned short* Bih   = (unsigned short*)carve((size_t)D * G * 2);
  unsigned short* Bhh   = (unsigned short*)carve((size_t)D * G * 2);
  float*          wn    = (float*)carve((size_t)NN * 4);
  float*          hg    = (float*)carve((size_t)D * 4);
  float*          denom = (float*)carve(4);
  float*          g     = (float*)carve((size_t)2048 * 4);
  float*          h1    = (float*)carve((size_t)512 * 4);

  // -------- pack weights into WMMA B-fragment bf16 layout --------
  {
    int tot = D * D;
    for (int l = 0; l < 4; ++l)
      pack_b_kernel<<<(tot + 255) / 256, 256, 0, stream>>>(
          ggnn_w + (size_t)l * D * D, BpG + (size_t)l * D * D, D, D, 0);
    int totG = D * G;
    pack_b_kernel<<<(totG + 255) / 256, 256, 0, stream>>>(gru_w_ih, Bih, D, G, 1);
    pack_b_kernel<<<(totG + 255) / 256, 256, 0, stream>>>(gru_w_hh, Bhh, D, G, 1);
  }

  // -------- embedding gather (f32 + bf16 mirror) --------
  embed_kernel<<<(NN * D + 255) / 256, 256, 0, stream>>>(x, embed_w, h, hb, NN);

  // -------- 4 GGNN layers --------
  for (int l = 0; l < 4; ++l) {
    dim3 gM(D / 64, (NN + 127) / 128);
    gemm_bf16_wmma<256, true><<<gM, 256, 0, stream>>>(hb, BpG + (size_t)l * D * D, m, NN, D);

    hipMemsetAsync(agg, 0, (size_t)NN * D * 4, stream);
    size_t sThreads = (size_t)NE * 64;
    scatter_add_kernel<<<(int)((sThreads + 255) / 256), 256, 0, stream>>>(
        m, edge_index, edge_index + NE, agg, NE);

    for (int c = 0; c < NN; c += CHUNK) {
      int mc = NN - c < CHUNK ? NN - c : CHUNK;
      dim3 gC(G / 64, (mc + 127) / 128);
      gemm_bf16_wmma<256, false><<<gC, 256, 0, stream>>>(agg + (size_t)c * D, Bih, gi, mc, G);
      gemm_bf16_wmma<256, true ><<<gC, 256, 0, stream>>>(hb  + (size_t)c * D, Bhh, gh, mc, G);
      gru_update_kernel<<<(mc * D + 255) / 256, 256, 0, stream>>>(
          gi, gh, gru_b_ih, gru_b_hh, h, hb, c, mc);
    }
  }

  // -------- gated pooling (softmax folded into weighted mean) --------
  hipMemsetAsync(denom, 0, 4, stream);
  hipMemsetAsync(hg, 0, D * 4, stream);
  gate_node_kernel<<<(NN + 7) / 8, 256, 0, stream>>>(h, gate_w, gate_b, wn, denom, NN);
  gate_accum_kernel<<<256, 256, 0, stream>>>(h, wn, hg, NN);

  // -------- LSTM step + logits --------
  lstm_matvec_kernel<<<256, 256, 0, stream>>>(lstm_w_ih, hg, denom, lstm_b_ih, lstm_b_hh, g);
  lstm_act_kernel<<<1, 512, 0, stream>>>(g, h1);
  dense_kernel<<<(NV + 7) / 8, 256, 0, stream>>>(dense_w, dense_b, h1, out, NV);
}
